// SNNClassifier_77841987272906
// MI455X (gfx1250) — matrix operand order repaired and checked by reading, not compiled, not run on hardware
//
#include <hip/hip_runtime.h>

// ---------------------------------------------------------------------------
// SNN 2-layer LIF forward, persistent over T=500 steps.
// T=500, B=1024, NIN=96, NH=128, NOUT=80. BETA=0.95, THRESH=1.0.
// Each block: 16 batch rows, 4 waves. Wave w computes layer-1 N-tiles {2w,2w+1}
// and layer-2 N-tile(s) {0,1}/{2}/{3}/{4}. Membrane state lives in f32 WMMA
// accumulator fragments for the whole time loop. W1/W2 cached in LDS as f16.
// ---------------------------------------------------------------------------

typedef _Float16 v16h __attribute__((ext_vector_type(16)));
typedef _Float16 v8h  __attribute__((ext_vector_type(8)));
typedef float    v8f  __attribute__((ext_vector_type(8)));
typedef float    v4f  __attribute__((ext_vector_type(4)));

#define WMMA_F16(a, b, c) \
  __builtin_amdgcn_wmma_f32_16x16x32_f16(false, (a), false, (b), (short)0, (c), false, false)

static __device__ __forceinline__ v16h cat16(v8h lo, v8h hi) {
  return __builtin_shufflevector(lo, hi, 0,1,2,3,4,5,6,7,8,9,10,11,12,13,14,15);
}

// A-fragment (16x32 f16) from global f32 row: halves 0..7 = cols c0..c0+7,
// halves 8..15 = cols c0+16..c0+23 (per-lane c0 already includes hi*8).
static __device__ __forceinline__ v16h load_a_global(const float* __restrict__ rp) {
  v4f f0 = *(const v4f*)(rp);
  v4f f1 = *(const v4f*)(rp + 4);
  v4f f2 = *(const v4f*)(rp + 16);
  v4f f3 = *(const v4f*)(rp + 20);
  v16h a;
  a[0]=(_Float16)f0[0];  a[1]=(_Float16)f0[1];  a[2]=(_Float16)f0[2];  a[3]=(_Float16)f0[3];
  a[4]=(_Float16)f1[0];  a[5]=(_Float16)f1[1];  a[6]=(_Float16)f1[2];  a[7]=(_Float16)f1[3];
  a[8]=(_Float16)f2[0];  a[9]=(_Float16)f2[1];  a[10]=(_Float16)f2[2]; a[11]=(_Float16)f2[3];
  a[12]=(_Float16)f3[0]; a[13]=(_Float16)f3[1]; a[14]=(_Float16)f3[2]; a[15]=(_Float16)f3[3];
  return a;
}

// B-fragment (32x16 f16): K offsets {0..7} and {16..23} rel. to p for column N.
static __device__ __forceinline__ v16h load_b_lds(const _Float16* p) {
  return cat16(*(const v8h*)p, *(const v8h*)(p + 16));
}
// A-fragment from LDS row-major f16: K offsets {0..7} and {16..23} rel. to p.
static __device__ __forceinline__ v16h load_a_lds(const _Float16* p) {
  return cat16(*(const v8h*)p, *(const v8h*)(p + 16));
}

__global__ __launch_bounds__(128)
void snn_lif2_kernel(const float* __restrict__ x,  const float* __restrict__ W1,
                     const float* __restrict__ b1, const float* __restrict__ W2,
                     const float* __restrict__ b2, float* __restrict__ out) {
  constexpr int   T    = 500;
  constexpr int   Bsz  = 1024;
  constexpr int   NIN  = 96;
  constexpr int   NH   = 128;
  constexpr int   NOUT = 80;
  constexpr int   W1S  = 104;  // padded row stride (halves): conflict-free b128 reads
  constexpr int   W2S  = 136;
  constexpr int   SPS  = 136;
  constexpr float BETA = 0.95f, TH = 1.0f;

  __shared__ alignas(16) _Float16 sW1[NH * W1S];     // 26.0 KB  W1 (n, k) f16
  __shared__ alignas(16) _Float16 sW2[NOUT * W2S];   // 21.3 KB  W2 (n, k) f16
  __shared__ alignas(16) _Float16 sSpk[16 * SPS];    //  4.3 KB  spk1 tile (row, k)

  const int tid  = threadIdx.x;
  const int lane = tid & 31;
  const int wave = tid >> 5;   // 0..3
  const int l15  = lane & 15;
  const int hi   = lane >> 4;  // 0|1 half-wave

  // Stage f32 weights -> f16 LDS (once).
  for (int i = tid; i < NH * NIN; i += 128)
    sW1[(i / NIN) * W1S + (i % NIN)] = (_Float16)W1[i];
  for (int i = tid; i < NOUT * NH; i += 128)
    sW2[(i / NH) * W2S + (i % NH)] = (_Float16)W2[i];
  __syncthreads();

  // Per-wave tile assignment + biases (per-lane column constants).
  const int   j0   = wave * 2;                       // layer-1 N-tiles j0, j0+1
  const float b1v0 = b1[j0 * 16 + l15];
  const float b1v1 = b1[j0 * 16 + 16 + l15];
  const int   jt2  = (wave == 0) ? 0 : (wave + 1);   // layer-2 N-tile (wave0 also does 1)
  const float b2v0 = b2[jt2 * 16 + l15];
  const float b2v1 = (wave == 0) ? b2[16 + l15] : 0.0f;

  // Membrane state in f32 accumulator fragments, persistent across T.
  v8f m1a = {}, m1b = {}, m2a = {}, m2b = {};

  const float*    xrow = x + (size_t)(blockIdx.x * 16 + l15) * NIN;
  float*          oblk = out + (size_t)blockIdx.x * 16 * NOUT;
  const size_t    TOT  = (size_t)T * Bsz * NOUT;     // mem_out offset in d_out
  const _Float16* wb1a = sW1 + (16 * j0 + l15) * W1S + hi * 8;
  const _Float16* wb1b = wb1a + 16 * W1S;
  const _Float16* wb2a = sW2 + (16 * jt2 + l15) * W2S + hi * 8;
  const _Float16* wb2b = sW2 + (16 * 1 + l15) * W2S + hi * 8;  // wave0 only
  const _Float16* sa   = sSpk + l15 * SPS + hi * 8;

  for (int t = 0; t < T; ++t) {
    // ---- load x_t A-fragments (16x96 f32 -> 3x f16 fragments) ----
    const int c0 = hi * 8;
    v16h a0 = load_a_global(xrow + c0);
    v16h a1 = load_a_global(xrow + 32 + c0);
    v16h a2 = load_a_global(xrow + 64 + c0);
    __builtin_prefetch(xrow + (size_t)Bsz * NIN, 0, 0);  // next timestep

    // ---- layer 1: cur1 = x W1^T ; LIF ; spk1 -> LDS ----
    {
      v8f c = {};
      c = WMMA_F16(a0, load_b_lds(wb1a), c);
      c = WMMA_F16(a1, load_b_lds(wb1a + 32), c);
      c = WMMA_F16(a2, load_b_lds(wb1a + 64), c);
#pragma unroll
      for (int e = 0; e < 8; ++e) {
        float mp = m1a[e];
        float r  = (mp > TH) ? TH : 0.0f;
        float mn = BETA * mp + c[e] + b1v0 - r;
        m1a[e]   = mn;
        sSpk[(e + hi * 8) * SPS + 16 * j0 + l15] = (_Float16)((mn > TH) ? 1.0f : 0.0f);
      }
    }
    {
      v8f c = {};
      c = WMMA_F16(a0, load_b_lds(wb1b), c);
      c = WMMA_F16(a1, load_b_lds(wb1b + 32), c);
      c = WMMA_F16(a2, load_b_lds(wb1b + 64), c);
#pragma unroll
      for (int e = 0; e < 8; ++e) {
        float mp = m1b[e];
        float r  = (mp > TH) ? TH : 0.0f;
        float mn = BETA * mp + c[e] + b1v1 - r;
        m1b[e]   = mn;
        sSpk[(e + hi * 8) * SPS + 16 * j0 + 16 + l15] = (_Float16)((mn > TH) ? 1.0f : 0.0f);
      }
    }
    __syncthreads();  // spk1 complete

    // ---- spk1 A-fragments (16x128 -> 4x f16 fragments) ----
    v16h s0 = load_a_lds(sa);
    v16h s1 = load_a_lds(sa + 32);
    v16h s2 = load_a_lds(sa + 64);
    v16h s3 = load_a_lds(sa + 96);
    __syncthreads();  // sSpk reads done; safe to overwrite next iteration

    // ---- layer 2: cur2 = spk1 W2^T ; LIF ; stream spk2/mem2 out ----
    float* orow = oblk + (size_t)t * Bsz * NOUT;
    {
      v8f c = {};
      c = WMMA_F16(s0, load_b_lds(wb2a), c);
      c = WMMA_F16(s1, load_b_lds(wb2a + 32), c);
      c = WMMA_F16(s2, load_b_lds(wb2a + 64), c);
      c = WMMA_F16(s3, load_b_lds(wb2a + 96), c);
#pragma unroll
      for (int e = 0; e < 8; ++e) {
        float  mp  = m2a[e];
        float  r   = (mp > TH) ? TH : 0.0f;
        float  mn  = BETA * mp + c[e] + b2v0 - r;
        m2a[e]     = mn;
        size_t off = (size_t)(e + hi * 8) * NOUT + jt2 * 16 + l15;
        __builtin_nontemporal_store((mn > TH) ? 1.0f : 0.0f, orow + off);
        __builtin_nontemporal_store(mn, orow + TOT + off);
      }
    }
    if (wave == 0) {  // second layer-2 tile (j2 = 1)
      v8f c = {};
      c = WMMA_F16(s0, load_b_lds(wb2b), c);
      c = WMMA_F16(s1, load_b_lds(wb2b + 32), c);
      c = WMMA_F16(s2, load_b_lds(wb2b + 64), c);
      c = WMMA_F16(s3, load_b_lds(wb2b + 96), c);
#pragma unroll
      for (int e = 0; e < 8; ++e) {
        float  mp  = m2b[e];
        float  r   = (mp > TH) ? TH : 0.0f;
        float  mn  = BETA * mp + c[e] + b2v1 - r;
        m2b[e]     = mn;
        size_t off = (size_t)(e + hi * 8) * NOUT + 16 + l15;
        __builtin_nontemporal_store((mn > TH) ? 1.0f : 0.0f, orow + off);
        __builtin_nontemporal_store(mn, orow + TOT + off);
      }
    }
    xrow += (size_t)Bsz * NIN;
  }
}

extern "C" void kernel_launch(void* const* d_in, const int* in_sizes, int n_in,
                              void* d_out, int out_size, void* d_ws, size_t ws_size,
                              hipStream_t stream) {
  (void)in_sizes; (void)n_in; (void)out_size; (void)d_ws; (void)ws_size;
  const float* x  = (const float*)d_in[0];
  const float* W1 = (const float*)d_in[1];
  const float* b1 = (const float*)d_in[2];
  const float* W2 = (const float*)d_in[3];
  const float* b2 = (const float*)d_in[4];
  float* out = (float*)d_out;

  dim3 grid(64);    // 1024 batch rows / 16 per block
  dim3 block(128);  // 4 wave32 waves
  hipLaunchKernelGGL(snn_lif2_kernel, grid, block, 0, stream, x, W1, b1, W2, b2, out);
}
